// GraphTM_63058709840610
// MI455X (gfx1250) — matrix operand order repaired and checked by reading, not compiled, not run on hardware
//
#include <hip/hip_runtime.h>
#include <hip/hip_bf16.h>

// ---------------------------------------------------------------------------
// GraphCast-style GNN for MI455X (gfx1250, wave32).
// All matmuls via v_wmma_f32_16x16x32_bf16 (bf16 A/B, f32 accumulate).
// One wave = one 16-row tile; 8 accumulator tiles cover N=128.
// B fragments of each K-chunk are preloaded as a group so the 16 b128 loads
// issue as one clause and only one wait precedes the 8 WMMAs.
// ---------------------------------------------------------------------------

#define H 128
#define LDSW 136          // padded LDS row stride (halves) to spread banks
#define LN_EPS 1e-5f

typedef __attribute__((ext_vector_type(16))) __bf16 v16bf;
typedef __attribute__((ext_vector_type(8)))  float  v8f;

__device__ __forceinline__ v8f wmma_bf16(v16bf a, v16bf b, v8f c) {
  return __builtin_amdgcn_wmma_f32_16x16x32_bf16(
      false, a, false, b, (short)0, c, false, false);
}

// Packed B fragment: frag (kc,nt) stored lane-major, 16 contiguous halves/lane.
__device__ __forceinline__ v16bf load_bfrag(const __bf16* wp, int kc, int nt, int lane) {
  return *(const v16bf*)(wp + (((size_t)(kc * 8 + nt)) * 32 + (size_t)lane) * 16);
}

// One K-chunk of a GEMM: preload all 8 B fragments, then 8 WMMAs.
__device__ __forceinline__ void gemm_chunk(v16bf a, const __bf16* wp, int kc,
                                           int lane, v8f acc[8]) {
  v16bf bfr[8];
#pragma unroll
  for (int nt = 0; nt < 8; ++nt) bfr[nt] = load_bfrag(wp, kc, nt, lane);
#pragma unroll
  for (int nt = 0; nt < 8; ++nt) acc[nt] = wmma_bf16(a, bfr[nt], acc[nt]);
}

// A fragment 16x32 bf16 from an f32 row. kb = kc*32 + ((lane>=16)?8:0).
__device__ __forceinline__ v16bf a_from_f32(const float* row, int kb) {
  v16bf a;
#pragma unroll
  for (int j = 0; j < 8; ++j) a[j] = (__bf16)row[kb + j];
#pragma unroll
  for (int j = 0; j < 8; ++j) a[8 + j] = (__bf16)row[kb + 16 + j];
  return a;
}

__device__ __forceinline__ v16bf a_from_f32_scaled(const float* row, int kb, float s) {
  v16bf a;
#pragma unroll
  for (int j = 0; j < 8; ++j) a[j]     = (__bf16)(row[kb + j] * s);
#pragma unroll
  for (int j = 0; j < 8; ++j) a[8 + j] = (__bf16)(row[kb + 16 + j] * s);
  return a;
}

// Second layer + LayerNorm. acc1 = x@W1 tiles (C layout). Result in acc2,
// already (y-mu)*rstd*g + b. Uses LDS to transpose SiLU(h) into A fragments.
__device__ __forceinline__ void mlp_tail(
    v8f acc1[8], const float* b1, const __bf16* w2p,
    const float* g, const float* bb, __bf16* hbuf, int lane, v8f acc2[8]) {
  const int cg = lane & 15;
  const int hi = lane >> 4;
  // bias + SiLU, scatter h tile to LDS (row-major, padded stride)
#pragma unroll
  for (int nt = 0; nt < 8; ++nt) {
    const int col = cg + nt * 16;
    const float bv = b1[col];
#pragma unroll
    for (int r = 0; r < 8; ++r) {
      float y = acc1[nt][r] + bv;
      float hval = y / (1.0f + __expf(-y));
      hbuf[(r + 8 * hi) * LDSW + col] = (__bf16)hval;
    }
  }
  __syncthreads();
  // second GEMM: K=128 -> 4 chunks
  const v8f vz = {0.f, 0.f, 0.f, 0.f, 0.f, 0.f, 0.f, 0.f};
#pragma unroll
  for (int i = 0; i < 8; ++i) acc2[i] = vz;
  const int khi = hi * 8;
  const __bf16* myrow = hbuf + (size_t)cg * LDSW;
#pragma unroll
  for (int kc = 0; kc < 4; ++kc) {
    v16bf a;
    const __bf16* p = myrow + kc * 32 + khi;
#pragma unroll
    for (int j = 0; j < 8; ++j) a[j] = p[j];
#pragma unroll
    for (int j = 0; j < 8; ++j) a[8 + j] = p[16 + j];
    gemm_chunk(a, w2p, kc, lane, acc2);
  }
  // LayerNorm per row (row r+8*hi lives in the 16-lane half this lane is in)
#pragma unroll
  for (int r = 0; r < 8; ++r) {
    float s1 = 0.f, s2 = 0.f;
#pragma unroll
    for (int nt = 0; nt < 8; ++nt) { float v = acc2[nt][r]; s1 += v; s2 += v * v; }
#pragma unroll
    for (int m = 1; m < 16; m <<= 1) { s1 += __shfl_xor(s1, m, 32); s2 += __shfl_xor(s2, m, 32); }
    float mean = s1 * (1.0f / H);
    float var = fmaxf(s2 * (1.0f / H) - mean * mean, 0.f);
    float rstd = rsqrtf(var + LN_EPS);
#pragma unroll
    for (int nt = 0; nt < 8; ++nt) acc2[nt][r] = (acc2[nt][r] - mean) * rstd;
  }
#pragma unroll
  for (int nt = 0; nt < 8; ++nt) {
    const int col = cg + nt * 16;
    const float gv = g[col], bv = bb[col];
#pragma unroll
    for (int r = 0; r < 8; ++r) acc2[nt][r] = acc2[nt][r] * gv + bv;
  }
}

// ------------------------------- kernels -----------------------------------

// Pack f32 KxN(=128) row-major weight into bf16 B fragments.
__global__ void __launch_bounds__(128) k_pack(const float* __restrict__ W,
                                              __bf16* __restrict__ out,
                                              int kin, int nfrag) {
  int fr = blockIdx.x * 4 + (threadIdx.x >> 5);
  int lane = threadIdx.x & 31;
  if (fr >= nfrag) return;
  int kc = fr >> 3, nt = fr & 7;
  int n = nt * 16 + (lane & 15);
  int kb = kc * 32 + (lane >> 4) * 8;
  __bf16* dst = out + ((size_t)fr * 32 + (size_t)lane) * 16;
#pragma unroll
  for (int j = 0; j < 16; ++j) {
    int k = kb + j + ((j >= 8) ? 8 : 0);
    dst[j] = (__bf16)((k < kin) ? W[(size_t)k * H + n] : 0.0f);
  }
}

__global__ void k_zero(float* p, size_t n) {
  size_t i = (size_t)blockIdx.x * blockDim.x + threadIdx.x;
  size_t st = (size_t)gridDim.x * blockDim.x;
  for (; i < n; i += st) p[i] = 0.f;
}

__global__ void k_count(const int* __restrict__ idx, int E, float* cnt) {
  int e = blockIdx.x * blockDim.x + threadIdx.x;
  if (e < E) atomicAdd(&cnt[idx[e]], 1.0f);
}

__global__ void k_inv(float* c, int n) {
  int i = blockIdx.x * blockDim.x + threadIdx.x;
  if (i < n) c[i] = 1.0f / fmaxf(c[i], 1.0f);
}

// Embed MLP: x [N,KIN] f32 -> out [N,128] f32. KIN compile-time so the padded
// A-fragment build is branch-free (no EXEC-masked per-element load loops).
template <int KIN>
__global__ void __launch_bounds__(128)
k_embed(const float* __restrict__ x,
        const __bf16* __restrict__ w1p, const float* __restrict__ b1,
        const __bf16* __restrict__ w2p, const float* __restrict__ g,
        const float* __restrict__ bb, float* __restrict__ out, int N) {
  constexpr int KC1 = (KIN + 31) / 32;
  __shared__ __bf16 lds[4][16 * LDSW];
  const int wid = threadIdx.x >> 5, lane = threadIdx.x & 31;
  const int tile = blockIdx.x * 4 + wid;
  const int row0 = tile * 16;
  int myrow = row0 + (lane & 15); if (myrow > N - 1) myrow = N - 1;
  const float* xr = x + (size_t)myrow * KIN;
  const int khi = (lane >> 4) * 8;
  const v8f vz = {0.f, 0.f, 0.f, 0.f, 0.f, 0.f, 0.f, 0.f};
  v8f acc1[8];
#pragma unroll
  for (int i = 0; i < 8; ++i) acc1[i] = vz;
#pragma unroll
  for (int kc = 0; kc < KC1; ++kc) {
    v16bf a;
#pragma unroll
    for (int j = 0; j < 8; ++j) {
      int k = kc * 32 + khi + j;
      a[j] = (k < KIN) ? (__bf16)xr[k] : (__bf16)0.0f;
    }
#pragma unroll
    for (int j = 0; j < 8; ++j) {
      int k = kc * 32 + khi + 16 + j;
      a[8 + j] = (k < KIN) ? (__bf16)xr[k] : (__bf16)0.0f;
    }
    gemm_chunk(a, w1p, kc, lane, acc1);
  }
  v8f acc2[8];
  mlp_tail(acc1, b1, w2p, g, bb, &lds[wid][0], lane, acc2);
  const int hi = lane >> 4, cg = lane & 15;
#pragma unroll
  for (int r = 0; r < 8; ++r) {
    int row = row0 + r + 8 * hi;
    if (row < N) {
#pragma unroll
      for (int nt = 0; nt < 8; ++nt)
        out[(size_t)row * H + cg + nt * 16] = acc2[nt][r];
    }
  }
}

// Edge MLP: eu = MLP([ea, xs[ids], xr[idr]]); ea += eu; agg[idr] += eu.
__global__ void __launch_bounds__(128)
k_edge(float* ea, const float* xs, const float* xr,
       const int* __restrict__ ids, const int* __restrict__ idr,
       const __bf16* __restrict__ w1p, const float* __restrict__ b1,
       const __bf16* __restrict__ w2p, const float* __restrict__ g,
       const float* __restrict__ bb, float* agg, int E) {
  __shared__ __bf16 lds[4][16 * LDSW];
  const int wid = threadIdx.x >> 5, lane = threadIdx.x & 31;
  const int tile = blockIdx.x * 4 + wid;
  const int row0 = tile * 16;
  int myrow = row0 + (lane & 15); if (myrow > E - 1) myrow = E - 1;
  const float* pe = ea + (size_t)myrow * H;
  const float* ps = xs + (size_t)ids[myrow] * H;
  const float* pr = xr + (size_t)idr[myrow] * H;
  const int khi = (lane >> 4) * 8;
  const v8f vz = {0.f, 0.f, 0.f, 0.f, 0.f, 0.f, 0.f, 0.f};
  v8f acc1[8];
#pragma unroll
  for (int i = 0; i < 8; ++i) acc1[i] = vz;
#pragma unroll
  for (int kc = 0; kc < 12; ++kc) {
    const float* src = (kc < 4) ? pe : ((kc < 8) ? ps : pr);
    v16bf a = a_from_f32(src, (kc & 3) * 32 + khi);
    gemm_chunk(a, w1p, kc, lane, acc1);
  }
  v8f acc2[8];
  mlp_tail(acc1, b1, w2p, g, bb, &lds[wid][0], lane, acc2);
  const int hi = lane >> 4, cg = lane & 15;
#pragma unroll
  for (int r = 0; r < 8; ++r) {
    int row = row0 + r + 8 * hi;
    if (row < E) {
      int ir = idr[row];
#pragma unroll
      for (int nt = 0; nt < 8; ++nt) {
        int col = cg + nt * 16;
        float eu = acc2[nt][r];
        size_t o = (size_t)row * H + col;
        ea[o] += eu;
        atomicAdd(&agg[(size_t)ir * H + col], eu);
      }
    }
  }
}

// Node MLP: nu = MLP([x, agg*inv]) (agg!=null) or MLP(x); out = x + nu.
// The agg branch is uniform (kernel argument), so __syncthreads stays legal.
__global__ void __launch_bounds__(128)
k_node(const float* x, const float* agg, const float* inv,
       const __bf16* __restrict__ w1p, const float* __restrict__ b1,
       const __bf16* __restrict__ w2p, const float* __restrict__ g,
       const float* __restrict__ bb, float* out, int N) {
  __shared__ __bf16 lds[4][16 * LDSW];
  const int wid = threadIdx.x >> 5, lane = threadIdx.x & 31;
  const int tile = blockIdx.x * 4 + wid;
  const int row0 = tile * 16;
  int myrow = row0 + (lane & 15); if (myrow > N - 1) myrow = N - 1;
  const float* px = x + (size_t)myrow * H;
  const int khi = (lane >> 4) * 8;
  const v8f vz = {0.f, 0.f, 0.f, 0.f, 0.f, 0.f, 0.f, 0.f};
  v8f acc1[8];
#pragma unroll
  for (int i = 0; i < 8; ++i) acc1[i] = vz;
#pragma unroll
  for (int kc = 0; kc < 4; ++kc)
    gemm_chunk(a_from_f32(px, kc * 32 + khi), w1p, kc, lane, acc1);
  if (agg) {
    const float* pa = agg + (size_t)myrow * H;
    const float sc = inv[myrow];
#pragma unroll
    for (int kc = 0; kc < 4; ++kc)
      gemm_chunk(a_from_f32_scaled(pa, kc * 32 + khi, sc), w1p, kc + 4, lane, acc1);
  }
  v8f acc2[8];
  mlp_tail(acc1, b1, w2p, g, bb, &lds[wid][0], lane, acc2);
  const int hi = lane >> 4, cg = lane & 15;
#pragma unroll
  for (int r = 0; r < 8; ++r) {
    int row = row0 + r + 8 * hi;
    if (row < N) {
#pragma unroll
      for (int nt = 0; nt < 8; ++nt) {
        int col = cg + nt * 16;
        size_t o = (size_t)row * H + col;
        out[o] = x[o] + acc2[nt][r];
      }
    }
  }
}

// ------------------------------- host --------------------------------------

struct MlpP { const float *b, *b1, *g, *w1, *w2; int kin; };
struct PackW { __bf16 *w1, *w2; int kc1; };

extern "C" void kernel_launch(void* const* d_in, const int* in_sizes, int n_in,
                              void* d_out, int out_size, void* d_ws, size_t ws_size,
                              hipStream_t stream) {
  const float* x_grid = (const float*)d_in[0];
  const float* x_mesh = (const float*)d_in[1];
  const float* g2m_ea = (const float*)d_in[2];
  const float* mm_ea  = (const float*)d_in[3];
  const float* m2g_ea = (const float*)d_in[4];
  const int* g2m_s = (const int*)d_in[5];
  const int* g2m_r = (const int*)d_in[6];
  const int* mm_s  = (const int*)d_in[7];
  const int* mm_r  = (const int*)d_in[8];
  const int* m2g_s = (const int*)d_in[9];
  const int* m2g_r = (const int*)d_in[10];

  const int NG = in_sizes[0] / 64;
  const int NM = in_sizes[1] / 3;
  const int EG2M = in_sizes[5];
  const int EMM  = in_sizes[7];
  const int EM2G = in_sizes[9];

  // Roles: 0..4 embeds (grid,mesh,g2m,mm,m2g), 5 enc.edge, 6 enc.recv,
  // 7 enc.send, 8..15 proc{0..3}.{edge,recv}, 16 dec.edge, 17 dec.recv.
  // Detect pytree flatten order: jax sorted-keys (leaf group = b,b1,g,w1,w2)
  // vs insertion order (w1,b1,w2,g,b).
  const bool sorted_keys = (in_sizes[11] == 128);
  static const int ORDER_SORTED[18] =
      {16, 17, 2, 0, 4, 1, 3, 5, 6, 7, 8, 9, 10, 11, 12, 13, 14, 15};
  MlpP mlps[18];
  for (int gi = 0; gi < 18; ++gi) {
    int base = 11 + 5 * gi;
    int role = sorted_keys ? ORDER_SORTED[gi] : gi;
    MlpP m;
    if (sorted_keys) {
      m.b  = (const float*)d_in[base + 0];
      m.b1 = (const float*)d_in[base + 1];
      m.g  = (const float*)d_in[base + 2];
      m.w1 = (const float*)d_in[base + 3];
      m.w2 = (const float*)d_in[base + 4];
      m.kin = in_sizes[base + 3] / H;
    } else {
      m.w1 = (const float*)d_in[base + 0];
      m.b1 = (const float*)d_in[base + 1];
      m.w2 = (const float*)d_in[base + 2];
      m.g  = (const float*)d_in[base + 3];
      m.b  = (const float*)d_in[base + 4];
      m.kin = in_sizes[base + 0] / H;
    }
    mlps[role] = m;
  }

  // ---- carve workspace ----
  char* wp = (char*)d_ws;
  auto alloc = [&](size_t bytes) -> void* {
    void* p = (void*)wp;
    wp += (bytes + 255) & ~(size_t)255;
    return p;
  };
  float* xg = (float*)alloc((size_t)NG * H * 4);
  float* xm = (float*)alloc((size_t)NM * H * 4);
  float* eg = (float*)alloc((size_t)EG2M * H * 4);
  float* em = (float*)alloc((size_t)EMM * H * 4);
  float* ed = (float*)alloc((size_t)EM2G * H * 4);
  const int NAGG = (NG > NM) ? NG : NM;
  float* agg = (float*)alloc((size_t)NAGG * H * 4);
  float* inv_g2m = (float*)alloc((size_t)NM * 4);
  float* inv_mm  = (float*)alloc((size_t)NM * 4);
  float* inv_m2g = (float*)alloc((size_t)NG * 4);

  // ---- pack weights into WMMA B-fragment order (bf16) ----
  PackW pw[18];
  for (int r = 0; r < 18; ++r) {
    int kc1 = (mlps[r].kin + 31) / 32;
    int nf1 = kc1 * 8;
    pw[r].kc1 = kc1;
    pw[r].w1 = (__bf16*)alloc((size_t)nf1 * 512 * 2);
    pw[r].w2 = (__bf16*)alloc((size_t)32 * 512 * 2);
    k_pack<<<(nf1 + 3) / 4, 128, 0, stream>>>(mlps[r].w1, pw[r].w1, mlps[r].kin, nf1);
    k_pack<<<8, 128, 0, stream>>>(mlps[r].w2, pw[r].w2, H, 32);
  }

  // ---- degree counts -> 1/max(cnt,1) (reused across processor layers) ----
  k_zero<<<256, 256, 0, stream>>>(inv_g2m, (size_t)NM);
  k_zero<<<256, 256, 0, stream>>>(inv_mm,  (size_t)NM);
  k_zero<<<256, 256, 0, stream>>>(inv_m2g, (size_t)NG);
  k_count<<<(EG2M + 255) / 256, 256, 0, stream>>>(g2m_r, EG2M, inv_g2m);
  k_count<<<(EMM  + 255) / 256, 256, 0, stream>>>(mm_r,  EMM,  inv_mm);
  k_count<<<(EM2G + 255) / 256, 256, 0, stream>>>(m2g_r, EM2G, inv_m2g);
  k_inv<<<(NM + 255) / 256, 256, 0, stream>>>(inv_g2m, NM);
  k_inv<<<(NM + 255) / 256, 256, 0, stream>>>(inv_mm,  NM);
  k_inv<<<(NG + 255) / 256, 256, 0, stream>>>(inv_m2g, NG);

  auto blocks16 = [](int n) { return ((n + 15) / 16 + 3) / 4; };

  // ---- embeds (KIN specialized at compile time) ----
  auto embed = [&](int r, const float* x, float* out, int N) {
    int kin = mlps[r].kin;
    if (kin == 64)
      k_embed<64><<<blocks16(N), 128, 0, stream>>>(
          x, pw[r].w1, mlps[r].b1, pw[r].w2, mlps[r].g, mlps[r].b, out, N);
    else if (kin == 4)
      k_embed<4><<<blocks16(N), 128, 0, stream>>>(
          x, pw[r].w1, mlps[r].b1, pw[r].w2, mlps[r].g, mlps[r].b, out, N);
    else
      k_embed<3><<<blocks16(N), 128, 0, stream>>>(
          x, pw[r].w1, mlps[r].b1, pw[r].w2, mlps[r].g, mlps[r].b, out, N);
  };
  embed(0, x_grid, xg, NG);
  embed(1, x_mesh, xm, NM);
  embed(2, g2m_ea, eg, EG2M);
  embed(3, mm_ea,  em, EMM);
  embed(4, m2g_ea, ed, EM2G);

  // ---- one message-passing round ----
  auto run_mp = [&](int re, int rr, float* ea, const float* xs, float* xr,
                    const int* ids, const int* idr, int E, int Nr,
                    const float* inv, float* out) {
    k_zero<<<2048, 256, 0, stream>>>(agg, (size_t)Nr * H);
    k_edge<<<blocks16(E), 128, 0, stream>>>(
        ea, xs, xr, ids, idr, pw[re].w1, mlps[re].b1, pw[re].w2,
        mlps[re].g, mlps[re].b, agg, E);
    k_node<<<blocks16(Nr), 128, 0, stream>>>(
        xr, agg, inv, pw[rr].w1, mlps[rr].b1, pw[rr].w2,
        mlps[rr].g, mlps[rr].b, out, Nr);
  };

  // encoder: grid -> mesh (then sender update on xg)
  run_mp(5, 6, eg, xg, xm, g2m_s, g2m_r, EG2M, NM, inv_g2m, xm);
  k_node<<<blocks16(NG), 128, 0, stream>>>(
      xg, (const float*)nullptr, (const float*)nullptr,
      pw[7].w1, mlps[7].b1, pw[7].w2, mlps[7].g, mlps[7].b, xg, NG);

  // processor: mesh -> mesh, 4 layers
  for (int i = 0; i < 4; ++i)
    run_mp(8 + 2 * i, 9 + 2 * i, em, xm, xm, mm_s, mm_r, EMM, NM, inv_mm, xm);

  // decoder: mesh -> grid, writes final x_grid to d_out
  run_mp(16, 17, ed, xm, xg, m2g_s, m2g_r, EM2G, NG, inv_m2g, (float*)d_out);
}
